// MeshConvPoint_62096637166349
// MI455X (gfx1250) — matrix operand order repaired and checked by reading, not compiled, not run on hardware
//
#include <hip/hip_runtime.h>
#include <stdint.h>

// Problem constants (match reference)
#define BB    4
#define CC    256
#define VV    32768
#define OO    512
#define KK    4
#define CKD   (CC * KK)     // 1024  (GEMM K dimension)

#define TILE_V 128
#define TILE_O 128
#define KSTEP  32           // ck per main-loop iteration (8 channels * 4 feats)
#define LSTRIDE 40          // ushorts per LDS row (80 B: 16B-aligned, bank-conflict-free)

typedef __attribute__((ext_vector_type(16))) __bf16 v16bf;
typedef __attribute__((ext_vector_type(8)))  float  v8f;
typedef __attribute__((ext_vector_type(4)))  unsigned int v4u;
typedef __attribute__((ext_vector_type(8)))  int    v8i;
typedef __attribute__((ext_vector_type(4)))  int    v4i;

union FragU { uint4 q[2]; v16bf v; };

__device__ __forceinline__ unsigned short f2bf(float f) {
  unsigned int u = __float_as_uint(f);
  u += 0x7FFFu + ((u >> 16) & 1u);   // round-to-nearest-even
  return (unsigned short)(u >> 16);
}

// ---------------------------------------------------------------------------
// Prep: one-time fp32 -> bf16 conversion of W into workspace (1 MB)
// ---------------------------------------------------------------------------
__global__ __launch_bounds__(256)
void convert_w_bf16(const float* __restrict__ W, unsigned int* __restrict__ Wbf) {
  const int i = blockIdx.x * 256 + threadIdx.x;       // packed-pair index
  const float2 w = ((const float2*)W)[i];
  Wbf[i] = (unsigned)f2bf(w.x) | ((unsigned)f2bf(w.y) << 16);
}

// ---------------------------------------------------------------------------
// Shared body: feature staging + fragment/WMMA math.
// A-tile (W, bf16) is staged by the caller-specific code.
// ---------------------------------------------------------------------------
__device__ __forceinline__ void stage_features(
    unsigned short* B_lds, const int4* idx_lds, const float* xb,
    int t, int c0) {
  #pragma unroll
  for (int i = 0; i < 4; ++i) {
    const int p  = t + i * 256;        // 1024 (v, channel) pairs
    const int v  = p & (TILE_V - 1);
    const int cl = p >> 7;             // 0..7 local channel
    const float* xr = xb + (size_t)(c0 + cl) * VV;
    const int4 g = idx_lds[v];
    const float f0 = xr[g.x];
    const float f1 = xr[g.y];
    const float f2 = xr[g.z];
    const float f3 = xr[g.w];
    const float s1  = f1 + f2 + f3;            // symm 1: sum
    const float p12 = f1 * f2;
    const float pr  = p12 * f3;                // symm 2: product
    const float s3  = p12 + f1 * f3 + f2 * f3; // symm 3: pairwise products
    const unsigned int u0 = (unsigned)f2bf(f0) | ((unsigned)f2bf(s1) << 16);
    const unsigned int u1 = (unsigned)f2bf(pr) | ((unsigned)f2bf(s3) << 16);
    *(uint2*)&B_lds[v * LSTRIDE + cl * 4] = make_uint2(u0, u1);
  }
}

__device__ __forceinline__ void frag_mma(
    const unsigned short* A_lds, const unsigned short* B_lds,
    v8f acc[2][4], int wo, int wv, int lrow, int lhalf) {
  FragU a[2];
  #pragma unroll
  for (int mo = 0; mo < 2; ++mo) {
    const int arow = wo * 32 + mo * 16 + lrow;
    const unsigned short* ap = &A_lds[arow * LSTRIDE + lhalf * 8];
    a[mo].q[0] = *(const uint4*)(ap);        // K = 8h .. 8h+7
    a[mo].q[1] = *(const uint4*)(ap + 16);   // K = 8h+16 .. 8h+23
  }
  #pragma unroll
  for (int nv = 0; nv < 4; ++nv) {
    const int col = wv * 64 + nv * 16 + lrow;
    const unsigned short* bp = &B_lds[col * LSTRIDE + lhalf * 8];
    FragU bf_;
    bf_.q[0] = *(const uint4*)(bp);          // K = 8h .. 8h+7
    bf_.q[1] = *(const uint4*)(bp + 16);     // K = 8h+16 .. 8h+23
    #pragma unroll
    for (int mo = 0; mo < 2; ++mo) {
      acc[mo][nv] = __builtin_amdgcn_wmma_f32_16x16x32_bf16(
          false, a[mo].v, false, bf_.v, (short)0, acc[mo][nv], false, false);
    }
  }
}

__device__ __forceinline__ void store_out(
    float* ob, const float* bias, v8f acc[2][4],
    int o0, int v0, int wo, int wv, int lrow, int lhalf) {
  #pragma unroll
  for (int mo = 0; mo < 2; ++mo) {
    const int obase = o0 + wo * 32 + mo * 16 + lhalf * 8;   // M = r + 8*lhalf
    #pragma unroll
    for (int nv = 0; nv < 4; ++nv) {
      const int col = v0 + wv * 64 + nv * 16 + lrow;        // N = lrow
      #pragma unroll
      for (int r = 0; r < 8; ++r) {
        const int o = obase + r;
        ob[(size_t)o * VV + col] = acc[mo][nv][r] + bias[o];
      }
    }
  }
}

// ---------------------------------------------------------------------------
// Main kernel, TDM variant: A-tile (bf16 W) copied by the Tensor Data Mover,
// overlapped with feature gather staging; tracked on TENSORcnt.
// ---------------------------------------------------------------------------
__global__ __launch_bounds__(256)
void meshconv_wmma_tdm(const float* __restrict__ x,
                       const int*   __restrict__ Gi,
                       const unsigned short* __restrict__ Wbf,
                       const float* __restrict__ bias,
                       float*       __restrict__ out) {
  __shared__ unsigned short A_lds[TILE_O * LSTRIDE];  // W tile   (bf16) 10 KB
  __shared__ unsigned short B_lds[TILE_V * LSTRIDE];  // features (bf16) 10 KB
  __shared__ int4 idx_lds[TILE_V];                    // gather indices   2 KB

  const int t     = threadIdx.x;
  const int lane  = t & 31;
  const int wave  = t >> 5;
  const int lrow  = lane & 15;
  const int lhalf = lane >> 4;
  const int wo    = wave & 3;
  const int wv    = wave >> 2;

  const int v0 = blockIdx.x * TILE_V;
  const int o0 = blockIdx.y * TILE_O;
  const int b  = blockIdx.z;

  if (t < TILE_V) {
    const int4* gp = (const int4*)(Gi + ((size_t)b * VV + (v0 + t)) * KK);
    idx_lds[t] = gp[0];
  }
  __syncthreads();

  v8f acc[2][4] = {};
  const float* xb = x + (size_t)b * CC * VV;

  // Invariant pieces of the tensor descriptor (CDNA5 ISA)
  const unsigned lds_base = (unsigned)(uintptr_t)A_lds;   // LDS aperture: low 32 bits = LDS byte addr
  // group1 word0: data_size=1 (2B) | pad_enable | pad_interval=3 (64B) | pad_amount=3 (16B)
  const unsigned g1w0 = (1u << 16) | (1u << 20) | (3u << 22) | (3u << 25);

  for (int it = 0; it < CKD / KSTEP; ++it) {
    const int ck0 = it * KSTEP;
    const int c0  = ck0 >> 2;

    __syncthreads();   // previous iteration's A/B reads complete

    if (wave == 0) {
      // --- issue TDM: 2D tile 32(ck) x 128(o) of bf16 W -> padded A_lds rows
      const unsigned long long ga =
          (unsigned long long)(uintptr_t)(Wbf + (size_t)o0 * CKD + ck0);
      v4u g0;
      g0[0] = 1u;                                  // count=1, user mode
      g0[1] = lds_base;                            // lds_addr (bytes)
      g0[2] = (unsigned)(ga & 0xFFFFFFFFu);        // global_addr[31:0]
      g0[3] = (unsigned)((ga >> 32) & 0x01FFFFFFu) // global_addr[56:32]
              | (2u << 30);                        // type=2 ("image")
      v8i g1;
      g1[0] = (int)g1w0;
      g1[1] = (int)((CKD & 0xFFFF) << 16);         // tensor_dim0[15:0] in [31:16]
      g1[2] = (int)(((CKD >> 16) & 0xFFFF)         // tensor_dim0[31:16]
              | ((OO & 0xFFFF) << 16));            // tensor_dim1[15:0]
      g1[3] = (int)(((OO >> 16) & 0xFFFF)          // tensor_dim1[31:16]
              | ((KSTEP & 0xFFFF) << 16));         // tile_dim0 = 32
      g1[4] = (int)(TILE_O & 0xFFFF);              // tile_dim1 = 128, tile_dim2 = 0
      g1[5] = (int)CKD;                            // tensor_dim0_stride[31:0] = 1024
      g1[6] = 0;                                   // stride[47:32] | dim1_stride[15:0]
      g1[7] = 0;
      v4i g2 = {0, 0, 0, 0};
      v4i g3 = {0, 0, 0, 0};
      v8i g4 = {0, 0, 0, 0, 0, 0, 0, 0};           // extra group (6-arg toolchain)
      __builtin_amdgcn_tensor_load_to_lds(g0, g1, g2, g3, g4, 0);
    }

    // --- feature staging overlaps the TDM copy ---
    stage_features(B_lds, idx_lds, xb, t, c0);

    if (wave == 0) {
      __builtin_amdgcn_s_wait_tensorcnt(0);        // A_lds tile landed
    }
    __syncthreads();

    frag_mma(A_lds, B_lds, acc, wo, wv, lrow, lhalf);
  }

  store_out(out + (size_t)b * OO * VV, bias, acc, o0, v0, wo, wv, lrow, lhalf);
}

// ---------------------------------------------------------------------------
// Fallback (no workspace): round-1 fused kernel, fp32 W converted per tile.
// ---------------------------------------------------------------------------
__global__ __launch_bounds__(256)
void meshconv_wmma_bf16(const float* __restrict__ x,
                        const int*   __restrict__ Gi,
                        const float* __restrict__ W,
                        const float* __restrict__ bias,
                        float*       __restrict__ out) {
  __shared__ unsigned short A_lds[TILE_O * LSTRIDE];
  __shared__ unsigned short B_lds[TILE_V * LSTRIDE];
  __shared__ int4 idx_lds[TILE_V];

  const int t     = threadIdx.x;
  const int lane  = t & 31;
  const int wave  = t >> 5;
  const int lrow  = lane & 15;
  const int lhalf = lane >> 4;
  const int wo    = wave & 3;
  const int wv    = wave >> 2;

  const int v0 = blockIdx.x * TILE_V;
  const int o0 = blockIdx.y * TILE_O;
  const int b  = blockIdx.z;

  if (t < TILE_V) {
    const int4* gp = (const int4*)(Gi + ((size_t)b * VV + (v0 + t)) * KK);
    idx_lds[t] = gp[0];
  }
  __syncthreads();

  v8f acc[2][4] = {};
  const float* xb = x + (size_t)b * CC * VV;

  for (int it = 0; it < CKD / KSTEP; ++it) {
    const int ck0 = it * KSTEP;
    const int c0  = ck0 >> 2;

    __syncthreads();

    {
      const int row = t >> 1, half = t & 1;
      const float4* wp = (const float4*)(W + (size_t)(o0 + row) * CKD + ck0 + half * 16);
      unsigned short* dst = &A_lds[row * LSTRIDE + half * 16];
      #pragma unroll
      for (int q = 0; q < 4; ++q) {
        float4 w4 = wp[q];
        dst[q * 4 + 0] = f2bf(w4.x);
        dst[q * 4 + 1] = f2bf(w4.y);
        dst[q * 4 + 2] = f2bf(w4.z);
        dst[q * 4 + 3] = f2bf(w4.w);
      }
    }

    stage_features(B_lds, idx_lds, xb, t, c0);

    __syncthreads();

    frag_mma(A_lds, B_lds, acc, wo, wv, lrow, lhalf);
  }

  store_out(out + (size_t)b * OO * VV, bias, acc, o0, v0, wo, wv, lrow, lhalf);
}

extern "C" void kernel_launch(void* const* d_in, const int* in_sizes, int n_in,
                              void* d_out, int out_size, void* d_ws, size_t ws_size,
                              hipStream_t stream) {
  (void)in_sizes; (void)n_in; (void)out_size;
  const float* x    = (const float*)d_in[0];
  const int*   Gi   = (const int*)d_in[1];
  const float* W    = (const float*)d_in[2];
  const float* bias = (const float*)d_in[3];
  float* out        = (float*)d_out;

  dim3 grid(VV / TILE_V, OO / TILE_O, BB);   // (256, 4, 4)
  const size_t wbf_bytes = (size_t)OO * CKD * sizeof(unsigned short);  // 1 MB

  if (ws_size >= wbf_bytes) {
    unsigned int* Wbf = (unsigned int*)d_ws;
    convert_w_bf16<<<(OO * CKD / 2) / 256, 256, 0, stream>>>(W, Wbf);
    meshconv_wmma_tdm<<<grid, 256, 0, stream>>>(
        x, Gi, (const unsigned short*)Wbf, bias, out);
  } else {
    meshconv_wmma_bf16<<<grid, 256, 0, stream>>>(x, Gi, W, bias, out);
  }
}
